// MultiHeadGlobalAttention_65240553226621
// MI455X (gfx1250) — matrix operand order repaired and checked by reading, not compiled, not run on hardware
//
#include <hip/hip_runtime.h>
#include <hip/hip_bf16.h>
#include <math.h>

// ---------------------------------------------------------------------------
// MI455X (gfx1250) multi-head attention, f16 WMMA path, f32 accumulate.
//   EMBED_DIM=1024, NUM_HEADS=16, HEAD_DIM=64, B=2, S=2048  (tokens M=4096)
// Stage 1: QKV GEMM   [4096,1024]x[1024,3072]  (f32 -> f16 out)
// Stage 2: flash attention, 2 waves/block, 64-key chunks, all-WMMA
// Stage 3: out  GEMM  [4096,1024]x[1024,1024]  (f16 -> f32 out + bias)
// ---------------------------------------------------------------------------

typedef _Float16 v16h __attribute__((ext_vector_type(16)));
typedef _Float16 v8h  __attribute__((ext_vector_type(8)));
typedef float    v8f  __attribute__((ext_vector_type(8)));
typedef float    v4f  __attribute__((ext_vector_type(4)));

static __device__ inline v8f zero8() {
  v8f z;
#pragma unroll
  for (int i = 0; i < 8; ++i) z[i] = 0.0f;
  return z;
}

static __device__ inline v16h mk16(v8h lo, v8h hi) {
  v16h r;
#pragma unroll
  for (int i = 0; i < 8; ++i) { r[i] = lo[i]; r[i + 8] = hi[i]; }
  return r;
}

static __device__ inline v16h wmma_a16(const _Float16* base, int half) {
  // A-fragment (16x32 f16): lane holds row M=lane&15, elements e -> k =
  // (e<8 ? e : e+8) + 8*half  => two contiguous 8-half (16B) runs.
  v8h lo = *reinterpret_cast<const v8h*>(base + 8 * half);
  v8h hi = *reinterpret_cast<const v8h*>(base + 16 + 8 * half);
  return mk16(lo, hi);
}

static __device__ inline v16h wmma_b16(const _Float16* base, int half) {
  // B-fragment (32x16 f16): lane holds col N=lane&15, elements e -> k =
  // (e<8 ? e : e+8) + 8*half  => two contiguous 8-half (16B) runs.
  v8h lo = *reinterpret_cast<const v8h*>(base + 8 * half);
  v8h hi = *reinterpret_cast<const v8h*>(base + 16 + 8 * half);
  return mk16(lo, hi);
}

// staging helpers: 8 contiguous elements -> 8 halves -----------------------
static __device__ inline v8h cvt8(const float* p) {
  v4f f0 = *reinterpret_cast<const v4f*>(p);
  v4f f1 = *reinterpret_cast<const v4f*>(p + 4);
  v8h h;
#pragma unroll
  for (int i = 0; i < 4; ++i) { h[i] = (_Float16)f0[i]; h[i + 4] = (_Float16)f1[i]; }
  return h;
}
static __device__ inline v8h cvt8(const _Float16* p) {
  return *reinterpret_cast<const v8h*>(p);
}
static __device__ inline void store_out(float* C, size_t idx, float v) { C[idx] = v; }
static __device__ inline void store_out(_Float16* C, size_t idx, float v) { C[idx] = (_Float16)v; }

// ---------------------------------------------------------------------------
// Tiled GEMM + bias:  C[M,N] = cast(A[M,K]) @ cast(Bw[K,N]) + bias[N]
// Block tile 64(M) x 128(N), BK=32, 256 threads = 8 waves (2x4).
// Each wave register-blocks a 32x32 output tile: 2 A-frags x 2 B-frags ->
// 4 v_wmma per K-step (fragments reused across the 2x2).
// B staged TRANSPOSED in LDS so B-fragments are contiguous ds_load_b128s.
// ---------------------------------------------------------------------------
template <typename AT, typename OT>
__global__ __launch_bounds__(256) void gemm_bias(const AT* __restrict__ A,
                                                 const float* __restrict__ Bw,
                                                 const float* __restrict__ bias,
                                                 OT* __restrict__ C,
                                                 int M, int N, int K) {
  __shared__ _Float16 As[64][40];    // [m][k]  row stride 80B (16B aligned)
  __shared__ _Float16 Bs[128][40];   // [n][k]  transposed

  const int tid   = threadIdx.x;
  const int tileN = blockIdx.x * 128;
  const int tileM = blockIdx.y * 64;
  const int wave  = tid >> 5;
  const int lane  = tid & 31;
  const int wm    = wave >> 2;       // 0..1  -> 32-row band
  const int wn    = wave & 3;        // 0..3  -> 32-col band
  const int half  = (lane >> 4) & 1;
  const int l16   = lane & 15;

  // A staging: 64x32 elems, 8/thread as one 16B run
  const int arow = tid >> 2;          // 0..63
  const int acol = (tid & 3) * 8;     // 0,8,16,24
  // B staging: 32x128 elems, 16/thread (one 64B run of n), scatter-transpose
  const int brow = tid >> 3;          // k row 0..31
  const int bcol = (tid & 7) * 16;    // 0..112

  v8f acc[4];
#pragma unroll
  for (int i = 0; i < 4; ++i) acc[i] = zero8();

  for (int k0 = 0; k0 < K; k0 += 32) {
    // --- stage A tile (64 x 32) ---
    {
      const AT* src = A + (size_t)(tileM + arow) * K + (k0 + acol);
      *reinterpret_cast<v8h*>(&As[arow][acol]) = cvt8(src);
    }
    // --- stage B tile (32k x 128n), transposed into Bs[n][k] ---
    {
      const float* src = Bw + (size_t)(k0 + brow) * N + (tileN + bcol);
#pragma unroll
      for (int q = 0; q < 4; ++q) {
        v4f f = *reinterpret_cast<const v4f*>(src + q * 4);
#pragma unroll
        for (int j = 0; j < 4; ++j) Bs[bcol + q * 4 + j][brow] = (_Float16)f[j];
      }
    }
    __syncthreads();

    // --- 2x2 register block: 4 WMMAs from 2 A-frags x 2 B-frags ---
    v16h af[2], bf[2];
#pragma unroll
    for (int i = 0; i < 2; ++i)
      af[i] = wmma_a16(&As[wm * 32 + i * 16 + l16][0], half);
#pragma unroll
    for (int j = 0; j < 2; ++j)
      bf[j] = wmma_b16(&Bs[wn * 32 + j * 16 + l16][0], half);
#pragma unroll
    for (int i = 0; i < 2; ++i)
#pragma unroll
      for (int j = 0; j < 2; ++j)
        acc[i * 2 + j] = __builtin_amdgcn_wmma_f32_16x16x32_f16(
            false, af[i], false, bf[j], (short)0, acc[i * 2 + j], false, false);
    __syncthreads();
  }

  // --- epilogue: bias + store (C layout: slot r -> row r+8*half, col l16) ---
#pragma unroll
  for (int i = 0; i < 2; ++i) {
#pragma unroll
    for (int j = 0; j < 2; ++j) {
      const int ncol  = tileN + wn * 32 + j * 16 + l16;
      const float b   = bias[ncol];
      const int mrow0 = tileM + wm * 32 + i * 16 + 8 * half;
      const v8f a = acc[i * 2 + j];
#pragma unroll
      for (int r = 0; r < 8; ++r)
        store_out(C, (size_t)(mrow0 + r) * N + ncol, a[r] + b);
    }
  }
}

// ---------------------------------------------------------------------------
// Flash attention: 2 waves/block, each wave owns 16 query rows; both waves
// share one 64-key K/V LDS stage per chunk (halves HBM traffic vs 1 wave).
// qkv layout per token: [h][{q|k|v}][64] f16 (stride 3072 per token).
// Per 64-key chunk: 8 WMMAs for Q@K^T (4 n-tiles x K=64) + 8 for P@V.
// ---------------------------------------------------------------------------
__global__ __launch_bounds__(64) void attention_kernel(const _Float16* __restrict__ qkv,
                                                       _Float16* __restrict__ attn) {
  __shared__ _Float16 Klds[64][64];     // [key][d]   row stride 128B
  __shared__ _Float16 Vt[64][80];       // [d][key]   row stride 160B (16B ok)
  __shared__ _Float16 Plds[2][16][64];  // per-wave [row][key], stride 128B

  const int tid  = threadIdx.x;
  const int wave = tid >> 5;
  const int lane = tid & 31;
  const int half = lane >> 4;
  const int l16  = lane & 15;
  const int h    = blockIdx.y;
  const int q0   = blockIdx.x * 32 + wave * 16;
  const size_t tok0 = (size_t)blockIdx.z * 2048;

  // Q fragments (16 rows x 64 d) -> two A-frags straight from global.
  const _Float16* qrow = qkv + (tok0 + q0 + l16) * 3072 + h * 192;
  const v16h qa0 = mk16(*reinterpret_cast<const v8h*>(qrow + 8 * half),
                        *reinterpret_cast<const v8h*>(qrow + 16 + 8 * half));
  const v16h qa1 = mk16(*reinterpret_cast<const v8h*>(qrow + 32 + 8 * half),
                        *reinterpret_cast<const v8h*>(qrow + 48 + 8 * half));

  v8f o[4];
#pragma unroll
  for (int d = 0; d < 4; ++d) o[d] = zero8();
  float mrow[8], lrow[8];
#pragma unroll
  for (int r = 0; r < 8; ++r) { mrow[r] = -1e30f; lrow[r] = 0.0f; }

  const float scale = 0.125f;  // 1/sqrt(64)

  for (int kc = 0; kc < 2048; kc += 64) {
    // --- stage K (row-major) + V (transposed); one key row per thread ---
    const _Float16* krow = qkv + (tok0 + kc + tid) * 3072 + h * 192 + 64;
    const _Float16* vrow = krow + 64;
    if (kc + 64 < 2048)
      __builtin_prefetch(krow + 64u * 3072u, 0, 0);  // global_prefetch_b8
#pragma unroll
    for (int i = 0; i < 8; ++i)
      *reinterpret_cast<v8h*>(&Klds[tid][i * 8]) =
          *reinterpret_cast<const v8h*>(krow + i * 8);
#pragma unroll
    for (int i = 0; i < 8; ++i) {
      v8h vv = *reinterpret_cast<const v8h*>(vrow + i * 8);
#pragma unroll
      for (int j = 0; j < 8; ++j) Vt[i * 8 + j][tid] = vv[j];
    }
    __syncthreads();

    // --- scores S = Q @ K^T : four 16x16 n-tiles, contraction K=64 ---
    v8f s[4];
#pragma unroll
    for (int t = 0; t < 4; ++t) {
      const _Float16* kb = &Klds[t * 16 + l16][0];
      v8f st = zero8();
      st = __builtin_amdgcn_wmma_f32_16x16x32_f16(false, qa0, false,
            wmma_b16(kb, half), (short)0, st, false, false);
      st = __builtin_amdgcn_wmma_f32_16x16x32_f16(false, qa1, false,
            wmma_b16(kb + 32, half), (short)0, st, false, false);
      s[t] = st;
    }

    // --- online softmax over 64 keys (row r+8*half lives in slot r) ---
    float alpha[8];
#pragma unroll
    for (int r = 0; r < 8; ++r) {
      float v0 = s[0][r] * scale, v1 = s[1][r] * scale;
      float v2 = s[2][r] * scale, v3 = s[3][r] * scale;
      float mx = fmaxf(fmaxf(v0, v1), fmaxf(v2, v3));
#pragma unroll
      for (int off = 8; off >= 1; off >>= 1)
        mx = fmaxf(mx, __shfl_xor(mx, off, 32));
      float mnew = fmaxf(mrow[r], mx);
      float p0 = expf(v0 - mnew), p1 = expf(v1 - mnew);
      float p2 = expf(v2 - mnew), p3 = expf(v3 - mnew);
      float rs = (p0 + p1) + (p2 + p3);
#pragma unroll
      for (int off = 8; off >= 1; off >>= 1)
        rs += __shfl_xor(rs, off, 32);
      alpha[r] = expf(mrow[r] - mnew);
      lrow[r] = lrow[r] * alpha[r] + rs;
      mrow[r] = mnew;
      _Float16* prow = &Plds[wave][r + 8 * half][0];
      prow[l16]      = (_Float16)p0;
      prow[16 + l16] = (_Float16)p1;
      prow[32 + l16] = (_Float16)p2;
      prow[48 + l16] = (_Float16)p3;
    }
#pragma unroll
    for (int d = 0; d < 4; ++d)
#pragma unroll
      for (int r = 0; r < 8; ++r) o[d][r] *= alpha[r];
    __syncthreads();

    // --- O += P @ V : P as two A-frags (keys 0..31 / 32..63) ---
    const v16h pa0 = wmma_a16(&Plds[wave][l16][0], half);
    const v16h pa1 = wmma_a16(&Plds[wave][l16][32], half);
#pragma unroll
    for (int d = 0; d < 4; ++d) {
      const _Float16* vb = &Vt[d * 16 + l16][0];
      o[d] = __builtin_amdgcn_wmma_f32_16x16x32_f16(false, pa0, false,
              wmma_b16(vb, half), (short)0, o[d], false, false);
      o[d] = __builtin_amdgcn_wmma_f32_16x16x32_f16(false, pa1, false,
              wmma_b16(vb + 32, half), (short)0, o[d], false, false);
    }
    __syncthreads();
  }

  // --- normalize and store f16: attn[token][h*64 + d] ---
#pragma unroll
  for (int r = 0; r < 8; ++r) {
    const float inv = 1.0f / lrow[r];
    const size_t tok = tok0 + q0 + r + 8 * half;
#pragma unroll
    for (int d = 0; d < 4; ++d) {
      attn[tok * 1024 + h * 64 + d * 16 + l16] = (_Float16)(o[d][r] * inv);
    }
  }
}

// ---------------------------------------------------------------------------
extern "C" void kernel_launch(void* const* d_in, const int* in_sizes, int n_in,
                              void* d_out, int out_size, void* d_ws, size_t ws_size,
                              hipStream_t stream) {
  const float* x     = (const float*)d_in[0];  // [2,2048,1024]
  const float* w_qkv = (const float*)d_in[1];  // [1024,3072]
  const float* b_qkv = (const float*)d_in[2];  // [3072]
  const float* w_o   = (const float*)d_in[3];  // [1024,1024]
  const float* b_o   = (const float*)d_in[4];  // [1024]
  float* out = (float*)d_out;                  // [2,2048,1024]

  const int M = 4096, D = 1024, N3 = 3072;

  _Float16* qkvh  = (_Float16*)d_ws;            // [4096,3072] f16
  _Float16* attnh = qkvh + (size_t)M * N3;      // [4096,1024] f16

  // 1) QKV projection (f32 in -> f16 out), 64x128 block tiles, 2x2 wmma/wave
  gemm_bias<float, _Float16><<<dim3(N3 / 128, M / 64), 256, 0, stream>>>(
      x, w_qkv, b_qkv, qkvh, M, N3, D);

  // 2) Flash attention: 2 waves/block share 64-key K/V stage, 32 queries/block
  attention_kernel<<<dim3(2048 / 32, 16, 2), 64, 0, stream>>>(qkvh, attnh);

  // 3) Output projection (f16 in -> f32 out) + bias
  gemm_bias<_Float16, float><<<dim3(D / 128, M / 64), 256, 0, stream>>>(
      attnh, w_o, b_o, out, M, D, D);
}
